// MinimalSuperPoint_4011499454522
// MI455X (gfx1250) — compile-verified, matrix operand order checked by reading, not run on hardware
//
#include <hip/hip_runtime.h>

// ---------------------------------------------------------------------------
// MinimalSuperPoint forward for MI455X (gfx1250).
// Channels-last activations, K-reordered weights [O][9][Cin] so every WMMA
// fragment is built from contiguous b128 loads. Input slabs are staged into
// LDS with global_load_async_to_lds_b128 (ASYNCcnt) and shared by all waves
// of a block. LDS column stride padded to Cin+8 halves (stride = 4 mod 32
// dwords -> 16 ds_load_b128 starts tile all 64 banks, conflict-free).
// Boundary taps redirect to a zeroed LDS region: branch-free inner loop.
// ---------------------------------------------------------------------------

typedef __attribute__((ext_vector_type(16))) _Float16 v16h;
typedef __attribute__((ext_vector_type(8)))  _Float16 v8h;
typedef __attribute__((ext_vector_type(8)))  float    v8f;

#define B_N 4
#define TOPK 1000

#define SHUF16(a, b) __builtin_shufflevector((a), (b), 0, 1, 2, 3, 4, 5, 6, 7, \
                                             8, 9, 10, 11, 12, 13, 14, 15)

static __device__ __forceinline__ void async_ld16(unsigned lds_off, const void* g) {
  unsigned long long ga = (unsigned long long)(uintptr_t)g;
  asm volatile("global_load_async_to_lds_b128 %0, %1, off"
               :: "v"(lds_off), "v"(ga) : "memory");
}

static __device__ __forceinline__ void wait_async0() {
#if __has_builtin(__builtin_amdgcn_s_wait_asynccnt)
  __builtin_amdgcn_s_wait_asynccnt(0);
#else
  asm volatile("s_wait_asynccnt 0x0" ::: "memory");
#endif
}

// ---------------------------------------------------------------------------
// Weight conversions.
// 3x3: f32 OIHW -> f16 [O][9][Cin]  (k = r*Cin + cin)
// 1x1: f32 [O][K] -> f16 [Opad][K], zero pad rows.
// ---------------------------------------------------------------------------
__global__ void cvt_w3_f16(const float* __restrict__ src, _Float16* __restrict__ dst,
                           int Cin, long long total) {
  long long i = (long long)blockIdx.x * blockDim.x + threadIdx.x;
  if (i >= total) return;
  int cin = (int)(i % Cin); long long t = i / Cin;
  int r = (int)(t % 9);     long long o = t / 9;
  dst[i] = (_Float16)src[((size_t)o * Cin + cin) * 9 + r];
}

__global__ void cvt_w1_f16(const float* __restrict__ src, _Float16* __restrict__ dst,
                           int O, int K, long long total) {
  long long i = (long long)blockIdx.x * blockDim.x + threadIdx.x;
  if (i >= total) return;
  long long o = i / K;
  dst[i] = (o < O) ? (_Float16)src[i] : (_Float16)0.f;
}

// ---------------------------------------------------------------------------
// e1: 1->64 3x3 conv + relu, direct, channels-last output [B][H][W][64].
// ---------------------------------------------------------------------------
__global__ void conv_e1(const float* __restrict__ img, const float* __restrict__ w,
                        const float* __restrict__ bias, _Float16* __restrict__ out,
                        int H, int W) {
  const long long HW = (long long)H * W;
  const long long tot = (long long)B_N * 64 * HW;
  long long i = (long long)blockIdx.x * blockDim.x + threadIdx.x;
  if (i >= tot) return;
  int oc = (int)(i % 64); long long t = i / 64;
  int x = (int)(t % W); t /= W;
  int y = (int)(t % H); int b = (int)(t / H);
  const float* im = img + (size_t)b * HW;
  float acc = bias[oc];
#pragma unroll
  for (int r = 0; r < 9; ++r) {
    int yy = y + r / 3 - 1, xx = x + r % 3 - 1;
    if (yy >= 0 && yy < H && xx >= 0 && xx < W)
      acc += w[oc * 9 + r] * im[(size_t)yy * W + xx];
  }
  out[i] = (_Float16)fmaxf(acc, 0.f);
}

// ---------------------------------------------------------------------------
// 3x3 conv + bias + relu. Channels-last f16 in/out, f32 accum via WMMA.
// Block = one 16-position tile (16 consecutive x in one row; W % 16 == 0)
// times ALL out-channels: blockDim = (32, Cout/16). Input slab 3x18 columns
// of Cin (stride Cin+8) async-staged into LDS; zeroed region for OOB taps.
// ---------------------------------------------------------------------------
__global__ void conv3x3_wmma(const _Float16* __restrict__ in,
                             const _Float16* __restrict__ wgt,
                             const float* __restrict__ bias,
                             _Float16* __restrict__ out,
                             int Cin, int Cout, int H, int W) {
  extern __shared__ _Float16 slab[];  // [3*18][Cin+8] + zreg[Cin+16]
  const int HW = H * W;
  const int cs = Cin + 8;             // padded column stride (halves)
  _Float16* zreg = slab + 54 * cs;
  const int nTile = blockIdx.x, b = blockIdx.y;
  const int lane = threadIdx.x, mTile = threadIdx.y;
  const int x0 = (nTile * 16) % W;
  const int y  = (nTile * 16) / W;
  const _Float16* __restrict__ inB = in + (size_t)b * HW * Cin;

  // ---- stage: zero region + async per-column b128 beats ----
  {
    const int t = mTile * 32 + lane;
    const int T = blockDim.y * 32;
    const v8h z = {0, 0, 0, 0, 0, 0, 0, 0};
    for (int i = t * 8; i < Cin + 16; i += T * 8)
      *(v8h*)(zreg + i) = z;                       // ds_store_b128
    const unsigned slabOff = (unsigned)(uintptr_t)(void*)slab;
    const int cb = Cin >> 3;                       // b128 beats per column
    for (int row = 0; row < 3; ++row) {
      const int yy = y + row - 1;
      if (yy < 0 || yy >= H) continue;
      for (int u = t; u < 18 * cb; u += T) {
        const int col = u / cb, bi = u - col * cb;
        const int xg = x0 - 1 + col;
        if (xg < 0 || xg >= W) continue;           // OOB col: never read
        const void* g = inB + ((size_t)yy * W + xg) * Cin + bi * 8;
        const unsigned l = slabOff + (unsigned)(((row * 18 + col) * cs + bi * 8) * 2);
        async_ld16(l, g);
      }
    }
  }
  wait_async0();
  __syncthreads();

  // ---- per-wave 16x16 tile GEMM (branch-free inner loop) ----
  const int nIdx = lane & 15;
  const int kH = (lane < 16) ? 0 : 8;       // K half-offset for this lane group
  const int oc = mTile * 16 + (lane & 15);  // A-matrix row
  const _Float16* __restrict__ wrow = wgt + (size_t)oc * Cin * 9;

  v8f acc = {0.f, 0.f, 0.f, 0.f, 0.f, 0.f, 0.f, 0.f};
#pragma unroll
  for (int r = 0; r < 9; ++r) {
    const int dr = r / 3, dc = r % 3;
    const int yy = y + dr - 1;
    const int xg = x0 + nIdx + dc - 1;
    const bool valid = (yy >= 0) && (yy < H) && (xg >= 0) && (xg < W);
    const _Float16* abase = wrow + r * Cin + kH;
    const _Float16* bbase = (valid ? slab + (dr * 18 + nIdx + dc) * cs : zreg) + kH;
    __builtin_prefetch(abase + Cin, 0, 0);
    for (int c0 = 0; c0 < Cin; c0 += 32) {
      v8h a0 = *(const v8h*)(abase + c0);
      v8h a1 = *(const v8h*)(abase + c0 + 16);
      v8h b0 = *(const v8h*)(bbase + c0);          // ds_load_b128, bank-tiled
      v8h b1 = *(const v8h*)(bbase + c0 + 16);
      v16h A = SHUF16(a0, a1);
      v16h Bf = SHUF16(b0, b1);
      acc = __builtin_amdgcn_wmma_f32_16x16x32_f16(false, A, false, Bf,
                                                   (short)0, acc, false, false);
    }
  }

  // ---- store: lane covers 8 contiguous out-channels at one position ----
  const int pos = nTile * 16 + nIdx;
  const int ocb = mTile * 16 + ((lane < 16) ? 0 : 8);
  v8h ov;
#pragma unroll
  for (int j = 0; j < 8; ++j)
    ov[j] = (_Float16)fmaxf(acc[j] + bias[ocb + j], 0.f);
  *(v8h*)(out + ((size_t)b * HW + pos) * Cout + ocb) = ov;   // b128 store
}

// ---------------------------------------------------------------------------
// 1x1 conv head: channels-last f16 in, padded channels-last f32 out
// [B][HW][Opad], no relu. Position tile async-staged to LDS (padded stride).
// blockDim = (32, Opad/16).
// ---------------------------------------------------------------------------
__global__ void conv1x1_wmma(const _Float16* __restrict__ in,
                             const _Float16* __restrict__ wgt,
                             const float* __restrict__ bias,
                             float* __restrict__ out,
                             int Cin, int Cout, int Opad, int HW) {
  extern __shared__ _Float16 slab[];  // [16][Cin+8]
  const int cs = Cin + 8;
  const int nTile = blockIdx.x, b = blockIdx.y;
  const int lane = threadIdx.x, mTile = threadIdx.y;
  const int pos0 = nTile * 16;
  const _Float16* __restrict__ inB = in + (size_t)b * HW * Cin;

  {
    const int t = mTile * 32 + lane;
    const int T = blockDim.y * 32;
    const unsigned slabOff = (unsigned)(uintptr_t)(void*)slab;
    const int cb = Cin >> 3;
    for (int u = t; u < 16 * cb; u += T) {
      const int col = u / cb, bi = u - col * cb;
      const void* g = inB + ((size_t)pos0 + col) * Cin + bi * 8;
      const unsigned l = slabOff + (unsigned)((col * cs + bi * 8) * 2);
      async_ld16(l, g);
    }
  }
  wait_async0();
  __syncthreads();

  const int nIdx = lane & 15;
  const int kH = (lane < 16) ? 0 : 8;
  const int oc = mTile * 16 + (lane & 15);
  const _Float16* __restrict__ wrow = wgt + (size_t)oc * Cin;
  const _Float16* __restrict__ bbase = slab + nIdx * cs + kH;

  v8f acc = {0.f, 0.f, 0.f, 0.f, 0.f, 0.f, 0.f, 0.f};
  for (int c0 = 0; c0 < Cin; c0 += 32) {
    v8h a0 = *(const v8h*)(wrow + kH + c0);
    v8h a1 = *(const v8h*)(wrow + kH + c0 + 16);
    v8h b0 = *(const v8h*)(bbase + c0);
    v8h b1 = *(const v8h*)(bbase + c0 + 16);
    v16h A = SHUF16(a0, a1);
    v16h Bf = SHUF16(b0, b1);
    acc = __builtin_amdgcn_wmma_f32_16x16x32_f16(false, A, false, Bf,
                                                 (short)0, acc, false, false);
  }

  const int pos = pos0 + nIdx;
  const int ocb = mTile * 16 + ((lane < 16) ? 0 : 8);
  float* op = out + ((size_t)b * HW + pos) * Opad + ocb;
  float4 q0, q1;
  q0.x = acc[0] + ((ocb + 0) < Cout ? bias[ocb + 0] : 0.f);
  q0.y = acc[1] + ((ocb + 1) < Cout ? bias[ocb + 1] : 0.f);
  q0.z = acc[2] + ((ocb + 2) < Cout ? bias[ocb + 2] : 0.f);
  q0.w = acc[3] + ((ocb + 3) < Cout ? bias[ocb + 3] : 0.f);
  q1.x = acc[4] + ((ocb + 4) < Cout ? bias[ocb + 4] : 0.f);
  q1.y = acc[5] + ((ocb + 5) < Cout ? bias[ocb + 5] : 0.f);
  q1.z = acc[6] + ((ocb + 6) < Cout ? bias[ocb + 6] : 0.f);
  q1.w = acc[7] + ((ocb + 7) < Cout ? bias[ocb + 7] : 0.f);
  *(float4*)op = q0;
  *(float4*)(op + 4) = q1;
}

// ---------------------------------------------------------------------------
// 2x2 max pool, stride 2, channels-last f16.
// ---------------------------------------------------------------------------
__global__ void maxpool2_f16(const _Float16* __restrict__ in, _Float16* __restrict__ out,
                             int C, int H, int W) {
  const int H2 = H >> 1, W2 = W >> 1;
  const long long tot = (long long)B_N * C * H2 * W2;
  long long i = (long long)blockIdx.x * blockDim.x + threadIdx.x;
  if (i >= tot) return;
  int c = (int)(i % C); long long t = i / C;
  int x = (int)(t % W2); t /= W2;
  int y = (int)(t % H2); int b = (int)(t / H2);
  const _Float16* p = in + (((size_t)b * H + 2 * y) * W + 2 * x) * C + c;
  const size_t rs = (size_t)W * C;
  float m = fmaxf(fmaxf((float)p[0], (float)p[C]),
                  fmaxf((float)p[rs], (float)p[rs + C]));
  out[i] = (_Float16)m;
}

// ---------------------------------------------------------------------------
// Softmax over 65 channels (padded stride 80), keep first 64.
// heat: [B][HW][80] f32 -> prob: [B][HW][64] f32
// ---------------------------------------------------------------------------
__global__ void softmax65(const float* __restrict__ heat, float* __restrict__ prob,
                          int HW) {
  long long i = (long long)blockIdx.x * blockDim.x + threadIdx.x;
  if (i >= (long long)B_N * HW) return;
  const float* h = heat + (size_t)i * 80;
  float mx = -3.4e38f;
  for (int c = 0; c < 65; ++c) mx = fmaxf(mx, h[c]);
  float s = 0.f;
  for (int c = 0; c < 65; ++c) s += __expf(h[c] - mx);
  float inv = 1.f / s;
  float* o = prob + (size_t)i * 64;
  for (int c = 0; c < 64; ++c) o[c] = __expf(h[c] - mx) * inv;
}

// ---------------------------------------------------------------------------
// 9x9 NMS + threshold. prob channels-last [B][HW][64];
// output channel-major [B][64][Hc][Wc] to match reference flat index order.
// ---------------------------------------------------------------------------
__global__ void nms_mask(const float* __restrict__ prob, float* __restrict__ outv,
                         int Hc, int Wc) {
  const long long tot = (long long)B_N * 64 * Hc * Wc;
  long long i = (long long)blockIdx.x * blockDim.x + threadIdx.x;
  if (i >= tot) return;
  int x = (int)(i % Wc); long long t = i / Wc;
  int y = (int)(t % Hc); t /= Hc;
  int c = (int)(t % 64); int b = (int)(t / 64);
  const float* m = prob + (size_t)b * Hc * Wc * 64;
  float h = m[((size_t)y * Wc + x) * 64 + c];
  float mp = -3.4e38f;
  for (int dy = -4; dy <= 4; ++dy) {
    int yy = y + dy; if (yy < 0 || yy >= Hc) continue;
    for (int dx = -4; dx <= 4; ++dx) {
      int xx = x + dx; if (xx < 0 || xx >= Wc) continue;
      mp = fmaxf(mp, m[((size_t)yy * Wc + xx) * 64 + c]);
    }
  }
  float k = (h == mp) ? h : 0.f;
  outv[i] = (k > 0.005f) ? k : 0.f;
}

// ---------------------------------------------------------------------------
// Per-batch top-k (k=1000 of N=307200): radix select on f32 bits (all >= 0),
// deterministic prefix-sum compaction, 1024-wide bitonic sort in LDS.
// ---------------------------------------------------------------------------
__global__ __launch_bounds__(1024) void topk_kernel(
    const float* __restrict__ vals, float* __restrict__ kpts,
    float* __restrict__ scores, int N, int Wc) {
  const int b = blockIdx.x;
  const int tid = threadIdx.x;
  const float* v = vals + (size_t)b * N;

  __shared__ unsigned hist[256];
  __shared__ unsigned s_prefix;
  __shared__ unsigned s_need;
  __shared__ unsigned cnt[1024];
  __shared__ float keys[1024];
  __shared__ int   ids[1024];

  if (tid == 0) { s_prefix = 0u; s_need = TOPK; }
  __syncthreads();

  for (int byte = 3; byte >= 0; --byte) {
    if (tid < 256) hist[tid] = 0u;
    __syncthreads();
    const unsigned pref = s_prefix;
    const int shift = (byte + 1) * 8;
    for (int i = tid; i < N; i += 1024) {
      unsigned bits = __float_as_uint(v[i]);
      bool match = (shift >= 32) || ((bits >> shift) == pref);
      if (match) atomicAdd(&hist[(bits >> (byte * 8)) & 255u], 1u);
    }
    __syncthreads();
    if (tid == 0) {
      unsigned cum = 0, need = s_need; int sel = 0;
      for (int bin = 255; bin >= 0; --bin) {
        unsigned c = hist[bin];
        if (cum + c >= need) { sel = bin; s_need = need - cum; break; }
        cum += c;
      }
      s_prefix = (s_prefix << 8) | (unsigned)sel;
    }
    __syncthreads();
  }
  const unsigned thr = s_prefix;

  keys[tid] = -1.f;
  ids[tid] = 0;
  unsigned c = 0;
  for (int i = tid; i < N; i += 1024)
    if (__float_as_uint(v[i]) >= thr) c++;
  cnt[tid] = c;
  __syncthreads();
  for (int off = 1; off < 1024; off <<= 1) {
    unsigned add = (tid >= off) ? cnt[tid - off] : 0u;
    __syncthreads();
    cnt[tid] += add;
    __syncthreads();
  }
  unsigned p = cnt[tid] - c;
  for (int i = tid; i < N; i += 1024) {
    if (__float_as_uint(v[i]) >= thr) {
      if (p < 1024) { keys[p] = v[i]; ids[p] = i; }
      p++;
    }
  }
  __syncthreads();

  for (int ksz = 2; ksz <= 1024; ksz <<= 1) {
    for (int j = ksz >> 1; j > 0; j >>= 1) {
      int i = tid, ixj = i ^ j;
      if (ixj > i) {
        bool desc = ((i & ksz) == 0);
        float a = keys[i], c2 = keys[ixj];
        int ia = ids[i], ic = ids[ixj];
        bool aBefore = (a > c2) || (a == c2 && ia <= ic);
        bool sw = desc ? !aBefore : aBefore;
        if (sw) { keys[i] = c2; keys[ixj] = a; ids[i] = ic; ids[ixj] = ia; }
      }
      __syncthreads();
    }
  }

  if (tid < TOPK) {
    float sc = keys[tid]; int id = ids[tid];
    if (sc < 0.f) { sc = 0.f; id = 0; }
    scores[b * TOPK + tid] = sc;
    float xi = (float)(id % Wc) * 8.f + 4.f;
    float yi = (float)(id / Wc) * 8.f + 4.f;
    kpts[((size_t)b * TOPK + tid) * 2 + 0] = xi;
    kpts[((size_t)b * TOPK + tid) * 2 + 1] = yi;
  }
}

// ---------------------------------------------------------------------------
// Bilinear descriptor sampling + L2 normalize. dmap channels-last
// [B][HW][256] so thread-per-channel taps are fully coalesced.
// ---------------------------------------------------------------------------
__global__ __launch_bounds__(256) void sample_desc(
    const float* __restrict__ dmap, const float* __restrict__ kpts,
    float* __restrict__ desc, int Hc, int Wc) {
  const int g = blockIdx.x;           // b*TOPK + kpt
  const int b = g / TOPK;
  const int c = threadIdx.x;
  const float xf = kpts[(size_t)g * 2 + 0];
  const float yf = kpts[(size_t)g * 2 + 1];
  const float px = xf / (float)(Wc * 8 - 1) * (float)(Wc - 1);
  const float py = yf / (float)(Hc * 8 - 1) * (float)(Hc - 1);
  const float x0f = floorf(px), y0f = floorf(py);
  const float wx = px - x0f, wy = py - y0f;
  const int x0 = (int)x0f, y0 = (int)y0f;
  const float* dm = dmap + (size_t)b * Hc * Wc * 256;

  auto tap = [&](int yy, int xx) -> float {
    bool valid = (xx >= 0) && (xx < Wc) && (yy >= 0) && (yy < Hc);
    int cy = min(max(yy, 0), Hc - 1), cx = min(max(xx, 0), Wc - 1);
    return valid ? dm[((size_t)cy * Wc + cx) * 256 + c] : 0.f;
  };
  float v = tap(y0, x0) * (1.f - wx) * (1.f - wy)
          + tap(y0, x0 + 1) * wx * (1.f - wy)
          + tap(y0 + 1, x0) * (1.f - wx) * wy
          + tap(y0 + 1, x0 + 1) * wx * wy;

  __shared__ float red[256];
  red[c] = v * v;
  __syncthreads();
  for (int s = 128; s > 0; s >>= 1) {
    if (c < s) red[c] += red[c + s];
    __syncthreads();
  }
  float n = fmaxf(sqrtf(red[0]), 1e-12f);
  desc[(size_t)g * 256 + c] = v / n;
}

// ---------------------------------------------------------------------------
// Launcher
// ---------------------------------------------------------------------------
extern "C" void kernel_launch(void* const* d_in, const int* in_sizes, int n_in,
                              void* d_out, int out_size, void* d_ws, size_t ws_size,
                              hipStream_t stream) {
  (void)in_sizes; (void)n_in; (void)out_size; (void)ws_size;

  const float* image = (const float*)d_in[0];
  const float *e1w = (const float*)d_in[1],  *e1b = (const float*)d_in[2];
  const float *e2w = (const float*)d_in[3],  *e2b = (const float*)d_in[4];
  const float *e3w = (const float*)d_in[5],  *e3b = (const float*)d_in[6];
  const float *e4w = (const float*)d_in[7],  *e4b = (const float*)d_in[8];
  const float *e5w = (const float*)d_in[9],  *e5b = (const float*)d_in[10];
  const float *e6w = (const float*)d_in[11], *e6b = (const float*)d_in[12];
  const float *e7w = (const float*)d_in[13], *e7b = (const float*)d_in[14];
  const float *e8w = (const float*)d_in[15], *e8b = (const float*)d_in[16];
  const float *d1w = (const float*)d_in[17], *d1b = (const float*)d_in[18];
  const float *d2w = (const float*)d_in[19], *d2b = (const float*)d_in[20];
  const float *s1w = (const float*)d_in[21], *s1b = (const float*)d_in[22];
  const float *s2w = (const float*)d_in[23], *s2b = (const float*)d_in[24];

  const int H0 = 480, W0 = 640;
  const int Hc = 60, Wc = 80, HWc = Hc * Wc;

  char* ws = (char*)d_ws;
  size_t off = 0;
  auto alloc = [&](size_t bytes) -> char* {
    char* p = ws + off;
    off += (bytes + 255) & ~(size_t)255;
    return p;
  };

  _Float16* bufA   = (_Float16*)alloc((size_t)B_N * 64 * H0 * W0 * 2);
  _Float16* bufB   = (_Float16*)alloc((size_t)B_N * 64 * H0 * W0 * 2);
  _Float16* headBf = (_Float16*)alloc((size_t)B_N * 256 * HWc * 2);
  float*    heat   = (float*)alloc((size_t)B_N * 80 * HWc * 4);   // padded 65->80
  float*    prob   = (float*)alloc((size_t)B_N * 64 * HWc * 4);
  float*    nmsv   = (float*)alloc((size_t)B_N * 64 * HWc * 4);
  float*    dmap   = (float*)alloc((size_t)B_N * 256 * HWc * 4);
  _Float16* wE2 = (_Float16*)alloc(64  * 576  * 2);
  _Float16* wE3 = (_Float16*)alloc(64  * 576  * 2);
  _Float16* wE4 = (_Float16*)alloc(64  * 576  * 2);
  _Float16* wE5 = (_Float16*)alloc(128 * 576  * 2);
  _Float16* wE6 = (_Float16*)alloc(128 * 1152 * 2);
  _Float16* wE7 = (_Float16*)alloc(128 * 1152 * 2);
  _Float16* wE8 = (_Float16*)alloc(128 * 1152 * 2);
  _Float16* wD1 = (_Float16*)alloc(256 * 1152 * 2);
  _Float16* wS1 = (_Float16*)alloc(256 * 1152 * 2);
  _Float16* wD2 = (_Float16*)alloc(80  * 256  * 2);
  _Float16* wS2 = (_Float16*)alloc(256 * 256  * 2);

  auto cvt3 = [&](const float* src, _Float16* dst, int O, int Cin) {
    long long tot = (long long)O * Cin * 9;
    cvt_w3_f16<<<dim3((unsigned)((tot + 255) / 256)), 256, 0, stream>>>(src, dst, Cin, tot);
  };
  auto cvt1 = [&](const float* src, _Float16* dst, int O, int K, int Opad) {
    long long tot = (long long)Opad * K;
    cvt_w1_f16<<<dim3((unsigned)((tot + 255) / 256)), 256, 0, stream>>>(src, dst, O, K, tot);
  };
  cvt3(e2w, wE2, 64, 64);   cvt3(e3w, wE3, 64, 64);
  cvt3(e4w, wE4, 64, 64);   cvt3(e5w, wE5, 128, 64);
  cvt3(e6w, wE6, 128, 128); cvt3(e7w, wE7, 128, 128);
  cvt3(e8w, wE8, 128, 128); cvt3(d1w, wD1, 256, 128);
  cvt3(s1w, wS1, 256, 128);
  cvt1(d2w, wD2, 65, 256, 80);
  cvt1(s2w, wS2, 256, 256, 256);

  // e1 direct conv (1->64), channels-last out
  {
    long long tot = (long long)B_N * 64 * H0 * W0;
    conv_e1<<<dim3((unsigned)((tot + 255) / 256)), 256, 0, stream>>>(image, e1w, e1b, bufA, H0, W0);
  }

  auto conv3 = [&](const _Float16* in, const _Float16* w, const float* bias,
                   _Float16* out, int Cin, int Cout, int H, int W) {
    dim3 g((unsigned)((H * W) / 16), B_N);
    dim3 blk(32, Cout / 16);
    size_t shm = (size_t)(54 * (Cin + 8) + Cin + 16) * 2;
    conv3x3_wmma<<<g, blk, shm, stream>>>(in, w, bias, out, Cin, Cout, H, W);
  };
  auto pool = [&](const _Float16* in, _Float16* out, int C, int H, int W) {
    long long tot = (long long)B_N * C * (H / 2) * (W / 2);
    maxpool2_f16<<<dim3((unsigned)((tot + 255) / 256)), 256, 0, stream>>>(in, out, C, H, W);
  };

  conv3(bufA, wE2, e2b, bufB, 64, 64, 480, 640);   // e2
  pool(bufB, bufA, 64, 480, 640);                  // -> 240x320
  conv3(bufA, wE3, e3b, bufB, 64, 64, 240, 320);   // e3
  conv3(bufB, wE4, e4b, bufA, 64, 64, 240, 320);   // e4
  pool(bufA, bufB, 64, 240, 320);                  // -> 120x160
  conv3(bufB, wE5, e5b, bufA, 64, 128, 120, 160);  // e5
  conv3(bufA, wE6, e6b, bufB, 128, 128, 120, 160); // e6
  pool(bufB, bufA, 128, 120, 160);                 // -> 60x80
  conv3(bufA, wE7, e7b, bufB, 128, 128, 60, 80);   // e7
  conv3(bufB, wE8, e8b, bufA, 128, 128, 60, 80);   // e8 -> feat in bufA

  auto conv1 = [&](const _Float16* in, const _Float16* w, const float* bias,
                   float* out, int Cin, int Cout, int Opad) {
    dim3 g((unsigned)(HWc / 16), B_N);
    dim3 blk(32, Opad / 16);
    size_t shm = (size_t)(16 * (Cin + 8) + Cin + 16) * 2;
    conv1x1_wmma<<<g, blk, shm, stream>>>(in, w, bias, out, Cin, Cout, Opad, HWc);
  };

  // Detector head
  conv3(bufA, wD1, d1b, headBf, 128, 256, 60, 80);
  conv1(headBf, wD2, d2b, heat, 256, 65, 80);
  // Descriptor head
  conv3(bufA, wS1, s1b, headBf, 128, 256, 60, 80);
  conv1(headBf, wS2, s2b, dmap, 256, 256, 256);

  {
    long long tot = (long long)B_N * HWc;
    softmax65<<<dim3((unsigned)((tot + 255) / 256)), 256, 0, stream>>>(heat, prob, HWc);
  }
  {
    long long tot = (long long)B_N * 64 * HWc;
    nms_mask<<<dim3((unsigned)((tot + 255) / 256)), 256, 0, stream>>>(prob, nmsv, Hc, Wc);
  }

  float* kpts   = (float*)d_out;                           // [4][1000][2]
  float* scores = (float*)d_out + (size_t)B_N * TOPK * 2;  // [4][1000]
  float* desc   = scores + (size_t)B_N * TOPK;             // [4][1000][256]

  topk_kernel<<<dim3(B_N), 1024, 0, stream>>>(nmsv, kpts, scores, 64 * HWc, Wc);
  sample_desc<<<dim3(B_N * TOPK), 256, 0, stream>>>(dmap, kpts, desc, Hc, Wc);
}
